// MinGRU_87308095193485
// MI455X (gfx1250) — matrix operand order repaired and checked by reading, not compiled, not run on hardware
//
#include <hip/hip_runtime.h>
#include <hip/hip_bf16.h>
#include <cmath>

// Problem dimensions (fixed by the reference)
#define B_  8
#define T_  4096
#define D_  1024
#define H_  1024
#define M_  (B_*T_)        // 32768 rows of x
#define NC  64             // scan chunks
#define TC  (T_/NC)        // 64 timesteps per chunk
#define BH  (B_*H_)        // 8192 channels

typedef __bf16 bf16;
typedef __attribute__((ext_vector_type(16))) __bf16 v16bf;
typedef __attribute__((ext_vector_type(8)))  __bf16 v8bf;
typedef __attribute__((ext_vector_type(8)))  float   v8f;

// Workspace layout (bytes)
#define OFF_XB 0UL                       // x in bf16: 32768*1024*2  = 64 MB
#define OFF_WZ 67108864UL                // Wz bf16:   1024*1024*2   =  2 MB
#define OFF_WH 69206016UL                // Wh bf16:                  =  2 MB
#define OFF_S  71303168UL                // s = log_z+log_g : f32    = 128 MB
#define OFF_L1 205520896UL               // log(1-z) : f32           = 128 MB
#define OFF_CA 339738624UL               // chunk cumsum partials    =  2 MB
#define OFF_CL 341835776UL               // chunk lse partials       =  2 MB

__device__ __forceinline__ float softplus_f(float x) {
    // log(1+e^x); stable for large |x|
    return (x > 15.0f) ? x : log1pf(expf(x));
}

__device__ __forceinline__ float logaddexp_f(float a, float b) {
    if (a == -INFINITY) return b;
    if (b == -INFINITY) return a;
    float m = fmaxf(a, b);
    return m + log1pf(expf(fminf(a, b) - m));
}

// ---------------------------------------------------------------------------
// f32 -> bf16 (round-to-nearest-even) conversion
// ---------------------------------------------------------------------------
__global__ void cvt_bf16_kernel(const float* __restrict__ in,
                                unsigned short* __restrict__ out, int n) {
    int i = blockIdx.x * blockDim.x + threadIdx.x;
    if (i < n) {
        unsigned int u = __float_as_uint(in[i]);
        u += 0x7FFFu + ((u >> 16) & 1u);
        out[i] = (unsigned short)(u >> 16);
    }
}

// ---------------------------------------------------------------------------
// Fused dual GEMM + activation epilogue.
// One wave computes a 32x32 tile of BOTH k = x*Wz^T + bz and th = x*Wh^T + bh,
// then writes s = log_z + log_g(th) and l1 = log(1-z).
// ---------------------------------------------------------------------------
__device__ __forceinline__ void emit_elem(float* __restrict__ s_out,
                                          float* __restrict__ l1_out,
                                          long m, long n, float k, float th) {
    float log_z   = -softplus_f(-k);                       // log sigmoid(k)
    float log_1mz = -softplus_f(k);                        // log(1 - sigmoid(k))
    float log_g   = (th >= 0.0f) ? logf(th + 0.5f)
                                 : -softplus_f(-th);       // Appendix B.3 log_g
    long idx = m * H_ + n;
    s_out[idx]  = log_z + log_g;
    l1_out[idx] = log_1mz;
}

__global__ __launch_bounds__(256)
void mingru_gemm_kernel(const unsigned short* __restrict__ xb_u,
                        const unsigned short* __restrict__ wzb_u,
                        const unsigned short* __restrict__ whb_u,
                        const float* __restrict__ bz,
                        const float* __restrict__ bh,
                        float* __restrict__ s_out,
                        float* __restrict__ l1_out) {
    const bf16* xb  = (const bf16*)xb_u;
    const bf16* wzb = (const bf16*)wzb_u;
    const bf16* whb = (const bf16*)whb_u;

    const int lane  = threadIdx.x & 31;
    const int wave  = (blockIdx.x * 256 + (int)threadIdx.x) >> 5;
    const int nTile = wave & (H_ / 32 - 1);   // 32 n-tiles
    const int mTile = wave >> 5;              // 1024 m-tiles
    const int m0 = mTile * 32;
    const int n0 = nTile * 32;

    const int lhalf = lane >> 4;              // 0..1
    const int lidx  = lane & 15;              // 0..15

    // A fragment (16x32 bf16): lane holds row m, contiguous K run [k0,k0+16)
    const long rowA0 = (long)(m0 + lidx) * D_;
    const long rowA1 = (long)(m0 + 16 + lidx) * D_;
    const int  akoff = lhalf * 16;
    // B fragment (32x16 bf16): lane holds col n, contiguous K run [k0,k0+16)
    const long colN0 = (long)(n0 + lidx) * D_;
    const long colN1 = (long)(n0 + 16 + lidx) * D_;
    const int  bkoff = lhalf * 16;

    v8f az00 = {}, az01 = {}, az10 = {}, az11 = {};
    v8f ah00 = {}, ah01 = {}, ah10 = {}, ah11 = {};

    for (int k = 0; k < D_; k += 32) {
        // A fragments for both m-halves (two 16B chunks each)
        v8bf a0lo = *(const v8bf*)(xb + rowA0 + k + akoff);
        v8bf a0hi = *(const v8bf*)(xb + rowA0 + k + akoff + 8);
        v8bf a1lo = *(const v8bf*)(xb + rowA1 + k + akoff);
        v8bf a1hi = *(const v8bf*)(xb + rowA1 + k + akoff + 8);
        v16bf a0 = __builtin_shufflevector(a0lo, a0hi,
                     0,1,2,3,4,5,6,7,8,9,10,11,12,13,14,15);
        v16bf a1 = __builtin_shufflevector(a1lo, a1hi,
                     0,1,2,3,4,5,6,7,8,9,10,11,12,13,14,15);
        // B fragments: contiguous 32B per lane
        v16bf bz0 = *(const v16bf*)(wzb + colN0 + k + bkoff);
        v16bf bz1 = *(const v16bf*)(wzb + colN1 + k + bkoff);
        v16bf bh0 = *(const v16bf*)(whb + colN0 + k + bkoff);
        v16bf bh1 = *(const v16bf*)(whb + colN1 + k + bkoff);

        az00 = __builtin_amdgcn_wmma_f32_16x16x32_bf16(false, a0, false, bz0, (short)0, az00, false, false);
        az01 = __builtin_amdgcn_wmma_f32_16x16x32_bf16(false, a0, false, bz1, (short)0, az01, false, false);
        az10 = __builtin_amdgcn_wmma_f32_16x16x32_bf16(false, a1, false, bz0, (short)0, az10, false, false);
        az11 = __builtin_amdgcn_wmma_f32_16x16x32_bf16(false, a1, false, bz1, (short)0, az11, false, false);
        ah00 = __builtin_amdgcn_wmma_f32_16x16x32_bf16(false, a0, false, bh0, (short)0, ah00, false, false);
        ah01 = __builtin_amdgcn_wmma_f32_16x16x32_bf16(false, a0, false, bh1, (short)0, ah01, false, false);
        ah10 = __builtin_amdgcn_wmma_f32_16x16x32_bf16(false, a1, false, bh0, (short)0, ah10, false, false);
        ah11 = __builtin_amdgcn_wmma_f32_16x16x32_bf16(false, a1, false, bh1, (short)0, ah11, false, false);
    }

    // Epilogue. C/D layout: lane holds col n = lidx (+16 for hi half tile),
    // VGPR r holds row m = 8*lhalf + r.
    const long nA = n0 + lidx;
    const long nB = n0 + 16 + lidx;
    const float bzA = bz[nA], bzB = bz[nB];
    const float bhA = bh[nA], bhB = bh[nB];

#pragma unroll
    for (int r = 0; r < 8; ++r) {
        long mA = m0 + lhalf * 8 + r;
        long mB = m0 + 16 + lhalf * 8 + r;
        emit_elem(s_out, l1_out, mA, nA, az00[r] + bzA, ah00[r] + bhA);
        emit_elem(s_out, l1_out, mA, nB, az01[r] + bzB, ah01[r] + bhB);
        emit_elem(s_out, l1_out, mB, nA, az10[r] + bzA, ah10[r] + bhA);
        emit_elem(s_out, l1_out, mB, nB, az11[r] + bzB, ah11[r] + bhB);
    }
}

// ---------------------------------------------------------------------------
// Chunked log-space scan. Pass 1: per (chan, chunk) local cumsum + local LSE.
// ---------------------------------------------------------------------------
__global__ __launch_bounds__(256)
void scan_pass1_kernel(const float* __restrict__ s, const float* __restrict__ l1,
                       float* __restrict__ cA, float* __restrict__ cL) {
    int tid  = blockIdx.x * 256 + (int)threadIdx.x;   // BH*NC threads
    int chan = tid & (BH - 1);
    int c    = tid >> 13;                              // BH = 2^13
    int b    = chan >> 10;                             // H = 2^10
    int h    = chan & (H_ - 1);
    long base = ((long)b * T_ + (long)c * TC) * H_ + h;

    float a = 0.0f, lse = -INFINITY;
    for (int t = 0; t < TC; ++t) {
        long idx = base + (long)t * H_;
        a += l1[idx];
        float v = s[idx] - a;
        lse = logaddexp_f(lse, v);
    }
    cA[tid] = a;      // layout [NC][BH]: tid == c*BH + chan
    cL[tid] = lse;
}

// Pass 2: exclusive combine over chunks per channel (in place).
__global__ __launch_bounds__(256)
void scan_pass2_kernel(float* __restrict__ cA, float* __restrict__ cL) {
    int chan = blockIdx.x * 256 + (int)threadIdx.x;   // BH threads
    float Ae = 0.0f, G = -INFINITY;
    for (int c = 0; c < NC; ++c) {
        long i = (long)c * BH + chan;
        float Ac = cA[i];
        float Lc = cL[i];
        cA[i] = Ae;
        cL[i] = G;
        G  = logaddexp_f(G, Lc - Ae);
        Ae += Ac;
    }
}

// Pass 3: replay local scan with exclusive prefixes, emit h = exp(log_x).
__global__ __launch_bounds__(256)
void scan_pass3_kernel(const float* __restrict__ s, const float* __restrict__ l1,
                       const float* __restrict__ cA, const float* __restrict__ cL,
                       float* __restrict__ out) {
    int tid  = blockIdx.x * 256 + (int)threadIdx.x;
    int chan = tid & (BH - 1);
    int c    = tid >> 13;
    int b    = chan >> 10;
    int h    = chan & (H_ - 1);
    long base = ((long)b * T_ + (long)c * TC) * H_ + h;

    const float Ae = cA[tid];
    const float G  = cL[tid];
    float a = 0.0f, lse = -INFINITY;
    for (int t = 0; t < TC; ++t) {
        long idx = base + (long)t * H_;
        a += l1[idx];
        float v = s[idx] - a;
        lse = logaddexp_f(lse, v);
        out[idx] = expf(Ae + a + logaddexp_f(G, lse - Ae));
    }
}

// ---------------------------------------------------------------------------
extern "C" void kernel_launch(void* const* d_in, const int* in_sizes, int n_in,
                              void* d_out, int out_size, void* d_ws, size_t ws_size,
                              hipStream_t stream) {
    (void)in_sizes; (void)n_in; (void)out_size; (void)ws_size;
    // setup_inputs order: x, Wz, bz, Wh, bh
    const float* x  = (const float*)d_in[0];
    const float* Wz = (const float*)d_in[1];
    const float* bz = (const float*)d_in[2];
    const float* Wh = (const float*)d_in[3];
    const float* bh = (const float*)d_in[4];
    float* out = (float*)d_out;

    char* w = (char*)d_ws;
    unsigned short* xb16  = (unsigned short*)(w + OFF_XB);
    unsigned short* wzb16 = (unsigned short*)(w + OFF_WZ);
    unsigned short* whb16 = (unsigned short*)(w + OFF_WH);
    float* s_buf  = (float*)(w + OFF_S);
    float* l1_buf = (float*)(w + OFF_L1);
    float* cA     = (float*)(w + OFF_CA);
    float* cL     = (float*)(w + OFF_CL);

    // 1) f32 -> bf16 conversions
    cvt_bf16_kernel<<<(M_ * D_) / 256, 256, 0, stream>>>(x,  xb16,  M_ * D_);
    cvt_bf16_kernel<<<(H_ * D_) / 256, 256, 0, stream>>>(Wz, wzb16, H_ * D_);
    cvt_bf16_kernel<<<(H_ * D_) / 256, 256, 0, stream>>>(Wh, whb16, H_ * D_);

    // 2) fused dual GEMM + activation epilogue
    //    waves = (M/32)*(H/32) = 32768; 8 waves per 256-thread block
    mingru_gemm_kernel<<<(M_ / 32) * (H_ / 32) / 8, 256, 0, stream>>>(
        xb16, wzb16, whb16, bz, bh, s_buf, l1_buf);

    // 3) chunked log-space scan
    scan_pass1_kernel<<<(BH * NC) / 256, 256, 0, stream>>>(s_buf, l1_buf, cA, cL);
    scan_pass2_kernel<<<BH / 256, 256, 0, stream>>>(cA, cL);
    scan_pass3_kernel<<<(BH * NC) / 256, 256, 0, stream>>>(s_buf, l1_buf, cA, cL, out);
}
